// Mitcls_CAM_unseg_35278861369901
// MI455X (gfx1250) — matrix operand order repaired and checked by reading, not compiled, not run on hardware
//
#include <hip/hip_runtime.h>
#include <hip/hip_bf16.h>
#include <stdint.h>

typedef __attribute__((ext_vector_type(16))) __bf16 v16bf;
typedef __attribute__((ext_vector_type(8)))  __bf16 v8bf;
typedef __attribute__((ext_vector_type(8)))  float  v8f;
typedef __attribute__((ext_vector_type(4))) unsigned int uint32x4;
typedef __attribute__((ext_vector_type(8))) int int32x8;
typedef __attribute__((ext_vector_type(4))) int int32x4;

__device__ __forceinline__ v16bf zero16() {
  v16bf z;
#pragma unroll
  for (int i = 0; i < 16; ++i) z[i] = (__bf16)0.0f;
  return z;
}

// ---------------------------------------------------------------------------
// Weight pre-pack into WMMA A-fragment order (16x32 A tile, contiguous k = 16*half + elem per lane, same as B)
// A2: [7 mtiles][36 ksteps][32 lanes][16 elems] bf16   (kstep = tap*4 + cchunk)
// A3: [2 mtiles][4  ksteps][32 lanes][16 elems] bf16
// ---------------------------------------------------------------------------
__global__ void prepack_kernel(const float* __restrict__ W2,
                               const float* __restrict__ W3,
                               __bf16* __restrict__ A2, __bf16* __restrict__ A3) {
  int idx = blockIdx.x * 256 + threadIdx.x;
  if (idx < 7 * 36 * 32 * 16) {
    int elem = idx & 15, lane = (idx >> 4) & 31;
    int kstep = (idx >> 9) % 36, mtile = (idx >> 9) / 36;
    int half = lane >> 4, m = lane & 15;
    int v = elem >> 1, pos = elem & 1;
    int koff = 16 * half + elem;
    int tap = kstep >> 2, cchunk = kstep & 3;
    int cin = cchunk * 32 + koff;
    int cout = mtile * 16 + m;
    int ky = tap / 3, kx = tap % 3;
    float val = 0.0f;
    if (cout < 100 && cin < 100)
      val = W2[((cout * 100 + cin) * 3 + ky) * 3 + kx];
    A2[idx] = (__bf16)val;
  }
  if (idx < 2 * 4 * 32 * 16) {
    int elem = idx & 15, lane = (idx >> 4) & 31;
    int kt = (idx >> 9) & 3, mtile = idx >> 11;
    int half = lane >> 4, m = lane & 15;
    int v = elem >> 1, pos = elem & 1;
    int koff = 16 * half + elem;
    int cin = kt * 32 + koff;
    int cout = mtile * 16 + m;
    float val = (cin < 100) ? W3[cout * 100 + cin] : 0.0f;
    A3[idx] = (__bf16)val;
  }
}

// Zero y2 pad channels (112..127) and the segment accumulators.
__global__ void init_kernel(__bf16* __restrict__ y2, float* __restrict__ sums,
                            float* __restrict__ cnts) {
  int p = blockIdx.x * 256 + threadIdx.x;  // 524288 threads
  if (p < 8 * 256 * 256) {
    v16bf z = zero16();
    *(v16bf*)(y2 + (size_t)p * 128 + 112) = z;
  }
  if (p < 8 * 32 * 20) sums[p] = 0.0f;
  if (p < 8 * 32) cnts[p] = 0.0f;
}

// 512x512 -> 256x256 bilinear (half-pixel) == exact 2x2 mean.
__global__ void resize_x_kernel(const float* __restrict__ x, float* __restrict__ xs) {
  int i = blockIdx.x * 256 + threadIdx.x;  // 8*3*256*256
  if (i >= 8 * 3 * 256 * 256) return;
  int xo = i & 255, yo = (i >> 8) & 255, plane = i >> 16;  // 24 planes
  const float* s = x + (size_t)plane * 512 * 512;
  int y2i = yo * 2, x2i = xo * 2;
  xs[i] = 0.25f * (s[y2i * 512 + x2i] + s[y2i * 512 + x2i + 1] +
                   s[(y2i + 1) * 512 + x2i] + s[(y2i + 1) * 512 + x2i + 1]);
}

// 1x1 classifier conv + spatial max (cls). One block per image.
__global__ void classifier_kernel(const float* __restrict__ x4,
                                  const float* __restrict__ Wc,
                                  float* __restrict__ featsmall,
                                  float* __restrict__ cls_out) {
  __shared__ float lw[20 * 512];
  __shared__ float red[256];
  int n = blockIdx.x, t = threadIdx.x;
  for (int i = t; i < 20 * 512; i += 256) lw[i] = Wc[i];
  __syncthreads();
  float acc[20];
#pragma unroll
  for (int c = 0; c < 20; ++c) acc[c] = 0.0f;
  const float* xp = x4 + (size_t)n * 512 * 256 + t;
  for (int k = 0; k < 512; ++k) {
    float xv = xp[(size_t)k * 256];
#pragma unroll
    for (int c = 0; c < 20; ++c) acc[c] += lw[c * 512 + k] * xv;
  }
  for (int c = 0; c < 20; ++c) featsmall[(n * 20 + c) * 256 + t] = acc[c];
  for (int c = 0; c < 20; ++c) {
    __syncthreads();
    red[t] = acc[c];
    __syncthreads();
    for (int s = 128; s > 0; s >>= 1) {
      if (t < s) red[t] = fmaxf(red[t], red[t + s]);
      __syncthreads();
    }
    if (t == 0) cls_out[n * 20 + c] = red[0];
  }
}

// conv1 3->100 3x3 + bias + ReLU + BN, writes y1 NHWC128 bf16 (pads zeroed).
__global__ void conv1_kernel(const float* __restrict__ xs, const float* __restrict__ W1,
                             const float* __restrict__ b1, const float* __restrict__ g1,
                             const float* __restrict__ be1, const float* __restrict__ m1,
                             const float* __restrict__ v1, __bf16* __restrict__ y1) {
  __shared__ float lw[100 * 27];
  int t = threadIdx.x;
  for (int i = t; i < 2700; i += 256) lw[i] = W1[i];
  __syncthreads();
  int blk = blockIdx.x;
  int xg = blk & 3, y = (blk >> 2) & 255, n = blk >> 10;
  int xo = xg * 64 + (t & 63);
  int cg = t >> 6;  // 4 groups of 32 channels
  float in[27];
  const float* xsp = xs + (size_t)n * 3 * 65536;
#pragma unroll
  for (int ci = 0; ci < 3; ++ci)
#pragma unroll
    for (int dy = 0; dy < 3; ++dy)
#pragma unroll
      for (int dx = 0; dx < 3; ++dx) {
        int sy = y + dy - 1, sx = xo + dx - 1;
        float v = ((unsigned)sy < 256u && (unsigned)sx < 256u)
                      ? xsp[ci * 65536 + sy * 256 + sx] : 0.0f;
        in[ci * 9 + dy * 3 + dx] = v;
      }
  __bf16* outp = y1 + ((size_t)(n * 256 + y) * 256 + xo) * 128;
  for (int j = 0; j < 32; ++j) {
    int c = cg * 32 + j;
    float o = 0.0f;
    if (c < 100) {
      float a = b1[c];
      const float* w = lw + c * 27;
#pragma unroll
      for (int k = 0; k < 27; ++k) a += w[k] * in[k];
      a = fmaxf(a, 0.0f);
      float s = g1[c] * rsqrtf(v1[c] + 1e-5f);
      o = (a - m1[c]) * s + be1[c];
    }
    outp[c] = (__bf16)o;
  }
}

// ---------------------------------------------------------------------------
// TDM: async-load one tap's worth of A fragments (2D tile: 2048 bf16 x 7 rows,
// row stride 18432 bf16) from global A2 into LDS. D# per ISA 8.3/8.4.
// ---------------------------------------------------------------------------
__device__ __forceinline__ void tdm_load_tap(const __bf16* A2, __bf16* ldsdst, int tap) {
  uint64_t ga = (uint64_t)(uintptr_t)(A2 + (size_t)tap * 2048);
  uint32_t lds = (uint32_t)(uintptr_t)ldsdst;
  uint32x4 g0;
  g0[0] = 1u;                                               // count=1, user D#
  g0[1] = lds;                                              // lds_addr
  g0[2] = (uint32_t)ga;                                     // global_addr lo
  g0[3] = (uint32_t)((ga >> 32) & 0x1FFFFFFu) | (2u << 30); // addr hi | type=2
  int32x8 g1;
  g1[0] = (int)(1u << 16);                 // data_size = 2 bytes
  g1[1] = (int)(18432u << 16);             // tensor_dim0[15:0]  (=18432)
  g1[2] = (int)((18432u >> 16) | (7u << 16)); // dim0 hi | tensor_dim1=7
  g1[3] = (int)(2048u << 16);              // tile_dim0 = 2048
  g1[4] = 7;                               // tile_dim1 = 7, tile_dim2 = 0
  g1[5] = 18432;                           // tensor_dim0_stride lo
  g1[6] = 0;
  g1[7] = 0;
  int32x4 gz;
  gz[0] = 0; gz[1] = 0; gz[2] = 0; gz[3] = 0;
#if defined(__clang_major__) && (__clang_major__ >= 23)
  int32x8 gz8;
#pragma unroll
  for (int i = 0; i < 8; ++i) gz8[i] = 0;
  __builtin_amdgcn_tensor_load_to_lds(g0, g1, gz, gz, gz8, 0);
#else
  __builtin_amdgcn_tensor_load_to_lds(g0, g1, gz, gz, 0);
#endif
}

// conv2 100->100 3x3 via implicit-GEMM WMMA bf16. Block=128 (4 waves), each wave
// does 112 out-ch x 16 spatial. A fragments staged to LDS by the Tensor Data
// Mover (double-buffered per 3x3 tap); all 7 A fragments are pulled into
// registers first so the 7 WMMAs issue back-to-back without per-op ds waits.
__global__ void __launch_bounds__(128)
conv2_kernel(const __bf16* __restrict__ y1, const __bf16* __restrict__ A2,
             const float* __restrict__ b2, const float* __restrict__ g2,
             const float* __restrict__ be2, const float* __restrict__ m2,
             const float* __restrict__ v2, __bf16* __restrict__ y2) {
  __shared__ __bf16 lA[2][7 * 2048];  // 2 x 28KB double buffer
  int blk = blockIdx.x;
  int xg = blk & 3, y = (blk >> 2) & 255, n = blk >> 10;
  int wave = threadIdx.x >> 5, lane = threadIdx.x & 31;
  int xbase = xg * 64 + wave * 16;
  int half = lane >> 4, nn = lane & 15;
  v8f acc[7];
#pragma unroll
  for (int mt = 0; mt < 7; ++mt)
#pragma unroll
    for (int r = 0; r < 8; ++r) acc[mt][r] = 0.0f;
  const __bf16* y1n = y1 + (size_t)n * 256 * 256 * 128;

  if (wave == 0) tdm_load_tap(A2, &lA[0][0], 0);

#pragma unroll 1
  for (int tap = 0; tap < 9; ++tap) {
    if (wave == 0) __builtin_amdgcn_s_wait_tensorcnt(0);
    __syncthreads();  // buffer[tap&1] ready; previous buffer free to refill
    if (wave == 0 && tap + 1 < 9) tdm_load_tap(A2, &lA[(tap + 1) & 1][0], tap + 1);

    int sy = y + tap / 3 - 1, sx = xbase + nn + tap % 3 - 1;
    bool inb = ((unsigned)sy < 256u) && ((unsigned)sx < 256u);
    const __bf16* bp = y1n + ((size_t)((sy & 255) * 256 + (sx & 255))) * 128 + 16 * half;
    // prefetch next tap's B row
    if (tap + 1 < 9) {
      int sy2 = y + (tap + 1) / 3 - 1, sx2 = xbase + nn + (tap + 1) % 3 - 1;
      if ((unsigned)sy2 < 256u && (unsigned)sx2 < 256u)
        __builtin_prefetch(y1n + ((size_t)(sy2 * 256 + sx2)) * 128 + 16 * half, 0, 0);
    }
    const __bf16* lbase = &lA[tap & 1][0] + lane * 16;
#pragma unroll
    for (int cc = 0; cc < 4; ++cc) {
      v16bf b;
      if (inb) b = *(const v16bf*)(bp + cc * 32);
      else b = zero16();
      // batch all A-fragment LDS loads, then issue WMMAs back-to-back
      v16bf a[7];
#pragma unroll
      for (int mt = 0; mt < 7; ++mt)
        a[mt] = *(const v16bf*)(lbase + mt * 2048 + cc * 512);
#pragma unroll
      for (int mt = 0; mt < 7; ++mt)
        acc[mt] = __builtin_amdgcn_wmma_f32_16x16x32_bf16(
            false, a[mt], false, b, (short)0, acc[mt], false, false);
    }
  }
  // epilogue: +bias, ReLU, BN -> bf16 NHWC128 (8-channel packed stores)
  int xo = xbase + nn;
  __bf16* outp = y2 + ((size_t)(n * 256 + y) * 256 + xo) * 128;
#pragma unroll
  for (int mt = 0; mt < 7; ++mt) {
    int c0 = mt * 16 + 8 * half;
    v8bf pk;
#pragma unroll
    for (int r = 0; r < 8; ++r) {
      int c = c0 + r;
      float o = 0.0f;
      if (c < 100) {
        float a = acc[mt][r] + b2[c];
        a = fmaxf(a, 0.0f);
        float s = g2[c] * rsqrtf(v2[c] + 1e-5f);
        o = (a - m2[c]) * s + be2[c];
      }
      pk[r] = (__bf16)o;
    }
    *(v8bf*)(outp + c0) = pk;
  }
}

// conv3 1x1 100->32 WMMA + bias + BN; writes mask (f32 NCHW) and per-pixel argmax.
__global__ void __launch_bounds__(128)
conv3_kernel(const __bf16* __restrict__ y2, const __bf16* __restrict__ A3,
             const float* __restrict__ b3, const float* __restrict__ g3,
             const float* __restrict__ be3, const float* __restrict__ m3,
             const float* __restrict__ v3, float* __restrict__ mask_out,
             int* __restrict__ maskobj) {
  int blk = blockIdx.x;
  int xg = blk & 3, y = (blk >> 2) & 255, n = blk >> 10;
  int wave = threadIdx.x >> 5, lane = threadIdx.x & 31;
  int xbase = xg * 64 + wave * 16;
  int half = lane >> 4, nn = lane & 15;
  v8f acc[2];
#pragma unroll
  for (int mt = 0; mt < 2; ++mt)
#pragma unroll
    for (int r = 0; r < 8; ++r) acc[mt][r] = 0.0f;
  const __bf16* y2p = y2 + ((size_t)(n * 256 + y) * 256 + xbase + nn) * 128;
#pragma unroll
  for (int kt = 0; kt < 4; ++kt) {
    v16bf b = *(const v16bf*)(y2p + kt * 32 + 16 * half);
    v16bf a[2];
#pragma unroll
    for (int mt = 0; mt < 2; ++mt)
      a[mt] = *(const v16bf*)(A3 + ((size_t)(mt * 4 + kt) * 32 + lane) * 16);
#pragma unroll
    for (int mt = 0; mt < 2; ++mt)
      acc[mt] = __builtin_amdgcn_wmma_f32_16x16x32_bf16(
          false, a[mt], false, b, (short)0, acc[mt], false, false);
  }
  int xo = xbase + nn;
  size_t sp = (size_t)y * 256 + xo;
  float best = -3.0e38f;
  int bestc = 0;
#pragma unroll
  for (int mt = 0; mt < 2; ++mt)
#pragma unroll
    for (int r = 0; r < 8; ++r) {
      int c = mt * 16 + 8 * half + r;  // strictly increasing per lane
      float s = g3[c] * rsqrtf(v3[c] + 1e-5f);
      float o = (acc[mt][r] + b3[c] - m3[c]) * s + be3[c];
      mask_out[(size_t)(n * 32 + c) * 65536 + sp] = o;
      if (o > best) { best = o; bestc = c; }
    }
  // combine with partner lane (other channel half), first-max tie-break
  float oval = __shfl_xor(best, 16, 32);
  int oc = __shfl_xor(bestc, 16, 32);
  if (oval > best || (oval == best && oc < bestc)) { best = oval; bestc = oc; }
  if (half == 0) maskobj[(size_t)n * 65536 + sp] = bestc;
}

// Segment sums: on-the-fly bilinear resample of 16x16 features, LDS accumulate.
__global__ void segsum_kernel(const float* __restrict__ featsmall,
                              const int* __restrict__ maskobj,
                              float* __restrict__ sums, float* __restrict__ cnts) {
  __shared__ float fs[20 * 256];
  __shared__ float lsum[32 * 20];
  __shared__ float lcnt[32];
  int n = blockIdx.x >> 6;
  int chunk = blockIdx.x & 63;
  int t = threadIdx.x;
  for (int i = t; i < 5120; i += 256) fs[i] = featsmall[n * 5120 + i];
  for (int i = t; i < 640; i += 256) lsum[i] = 0.0f;
  if (t < 32) lcnt[t] = 0.0f;
  __syncthreads();
  for (int it = 0; it < 4; ++it) {
    int p = chunk * 1024 + it * 256 + t;
    int id = maskobj[(size_t)n * 65536 + p];
    int yy = p >> 8, xx = p & 255;
    float sy = (yy + 0.5f) * 0.0625f - 0.5f;
    float sx = (xx + 0.5f) * 0.0625f - 0.5f;
    float fy = floorf(sy), fx = floorf(sx);
    float wy = sy - fy, wx = sx - fx;
    int iy0 = min(15, max(0, (int)fy));
    int iy1 = min(15, max(0, (int)fy + 1));
    int ix0 = min(15, max(0, (int)fx));
    int ix1 = min(15, max(0, (int)fx + 1));
    atomicAdd(&lcnt[id], 1.0f);
#pragma unroll
    for (int c = 0; c < 20; ++c) {
      const float* f = fs + c * 256;
      float v = (1.0f - wy) * ((1.0f - wx) * f[iy0 * 16 + ix0] + wx * f[iy0 * 16 + ix1]) +
                wy * ((1.0f - wx) * f[iy1 * 16 + ix0] + wx * f[iy1 * 16 + ix1]);
      atomicAdd(&lsum[id * 20 + c], v);
    }
  }
  __syncthreads();
  for (int i = t; i < 640; i += 256) atomicAdd(&sums[n * 640 + i], lsum[i]);
  if (t < 32) atomicAdd(&cnts[n * 32 + t], lcnt[t]);
}

__global__ void finalize_kernel(const float* __restrict__ sums,
                                const float* __restrict__ cnts,
                                float* __restrict__ means,
                                float* __restrict__ clsfea_out) {
  __shared__ float lm[640];
  __shared__ float lc[32];
  int n = blockIdx.x, t = threadIdx.x;  // 64 threads
  if (t < 32) lc[t] = cnts[n * 32 + t];
  __syncthreads();
  for (int i = t; i < 640; i += 64) {
    int s = i / 20;
    lm[i] = sums[n * 640 + i] / fmaxf(lc[s], 1.0f);
  }
  __syncthreads();
  for (int i = t; i < 640; i += 64) means[n * 640 + i] = lm[i];
  if (t < 20) {
    float best = -3.0e38f;
    for (int s = 0; s < 32; ++s)
      if (lc[s] > 0.0f) best = fmaxf(best, lm[s * 20 + t]);
    clsfea_out[n * 20 + t] = best;
  }
}

__global__ void scatter_kernel(const float* __restrict__ means,
                               const int* __restrict__ maskobj,
                               float* __restrict__ feat_out) {
  __shared__ float lm[640];
  int gp = blockIdx.x * 256 + threadIdx.x;  // 524288
  int n = gp >> 16;
  for (int i = threadIdx.x; i < 640; i += 256) lm[i] = means[n * 640 + i];
  __syncthreads();
  int id = maskobj[gp];
  int p = gp & 65535;
  float* o = feat_out + (size_t)n * 20 * 65536 + p;
#pragma unroll
  for (int c = 0; c < 20; ++c) o[(size_t)c * 65536] = lm[id * 20 + c];
}

// ---------------------------------------------------------------------------
extern "C" void kernel_launch(void* const* d_in, const int* in_sizes, int n_in,
                              void* d_out, int out_size, void* d_ws, size_t ws_size,
                              hipStream_t stream) {
  const float* x   = (const float*)d_in[0];
  const float* x4  = (const float*)d_in[1];
  const float* Wc  = (const float*)d_in[2];
  const float* W1  = (const float*)d_in[3];
  const float* b1  = (const float*)d_in[4];
  const float* g1  = (const float*)d_in[5];
  const float* be1 = (const float*)d_in[6];
  const float* m1  = (const float*)d_in[7];
  const float* v1  = (const float*)d_in[8];
  const float* W2  = (const float*)d_in[9];
  const float* b2  = (const float*)d_in[10];
  const float* g2  = (const float*)d_in[11];
  const float* be2 = (const float*)d_in[12];
  const float* m2  = (const float*)d_in[13];
  const float* v2  = (const float*)d_in[14];
  const float* W3  = (const float*)d_in[15];
  const float* b3  = (const float*)d_in[16];
  const float* g3  = (const float*)d_in[17];
  const float* be3 = (const float*)d_in[18];
  const float* m3  = (const float*)d_in[19];
  const float* v3  = (const float*)d_in[20];

  float* out = (float*)d_out;
  float* cls_out    = out;                      // [8,20]
  float* clsfea_out = out + 160;                // [8,20]
  float* feat_out   = out + 320;                // [8,20,256,256]
  float* mask_out   = out + 320 + 10485760;     // [8,32,256,256]

  // workspace layout (256B aligned offsets)
  char* ws = (char*)d_ws;
  size_t off = 0;
  auto take = [&](size_t bytes) { char* p = ws + off; off = (off + bytes + 255) & ~(size_t)255; return p; };
  float*  xs        = (float*)take((size_t)8 * 3 * 65536 * 4);
  __bf16* y1        = (__bf16*)take((size_t)8 * 65536 * 128 * 2);
  __bf16* y2        = (__bf16*)take((size_t)8 * 65536 * 128 * 2);
  __bf16* A2        = (__bf16*)take((size_t)7 * 36 * 32 * 16 * 2);
  __bf16* A3        = (__bf16*)take((size_t)2 * 4 * 32 * 16 * 2);
  float*  featsmall = (float*)take((size_t)8 * 20 * 256 * 4);
  int*    mobj      = (int*)take((size_t)8 * 65536 * 4);
  float*  sums      = (float*)take((size_t)8 * 32 * 20 * 4);
  float*  cnts      = (float*)take((size_t)8 * 32 * 4);
  float*  means     = (float*)take((size_t)8 * 32 * 20 * 4);
  (void)ws_size; (void)in_sizes; (void)n_in; (void)out_size;

  prepack_kernel<<<504, 256, 0, stream>>>(W2, W3, A2, A3);
  init_kernel<<<2048, 256, 0, stream>>>(y2, sums, cnts);
  resize_x_kernel<<<6144, 256, 0, stream>>>(x, xs);
  classifier_kernel<<<8, 256, 0, stream>>>(x4, Wc, featsmall, cls_out);
  conv1_kernel<<<8192, 256, 0, stream>>>(xs, W1, b1, g1, be1, m1, v1, y1);
  conv2_kernel<<<8192, 128, 0, stream>>>(y1, A2, b2, g2, be2, m2, v2, y2);
  conv3_kernel<<<8192, 128, 0, stream>>>(y2, A3, b3, g3, be3, m3, v3, mask_out, mobj);
  segsum_kernel<<<512, 256, 0, stream>>>(featsmall, mobj, sums, cnts);
  finalize_kernel<<<8, 64, 0, stream>>>(sums, cnts, means, clsfea_out);
  scatter_kernel<<<2048, 256, 0, stream>>>(means, mobj, feat_out);
}
